// Self_Attention_16114717295009
// MI455X (gfx1250) — matrix operand order repaired and checked by reading, not compiled, not run on hardware
//
#include <hip/hip_runtime.h>
#include <hip/hip_bf16.h>

typedef __attribute__((ext_vector_type(16))) __bf16 v16bf;
typedef __attribute__((ext_vector_type(8)))  __bf16 v8bf;
typedef __attribute__((ext_vector_type(8)))  float  v8f;

#define BATCH 4
#define SEQ   2048
#define DIM   1024

__device__ __forceinline__ void split_bf16(float f, __bf16& h, __bf16& l) {
    h = (__bf16)f;
    l = (__bf16)(f - (float)h);
}

// A-fragment (16x32 bf16, row-major source). p = &A[(row0+r)*lda + k0 + 16*half]
// Layout: lanes 0-15 row r hold K 0..15 (elems 0-15);
// lanes 16-31 hold K 16..31 -> handled by the 16*half offset (matches B).
__device__ __forceinline__ v16bf load_a_frag(const __bf16* p) {
    v8bf a0 = *(const v8bf*)(p);
    v8bf a1 = *(const v8bf*)(p + 8);
    v16bf r;
#pragma unroll
    for (int i = 0; i < 8; ++i) { r[i] = a0[i]; r[i + 8] = a1[i]; }
    return r;
}

// Generic bf16x3 split-precision GEMM: C = A * B, A [M x K] via (Ah,Al) row-major,
// B given transposed: Bt [N x K] via (Bh,Bl) row-major (so B(k,n) = Bt[n][k]).
// Wave tile: 32 (M) x 64 (N) -> 8 accumulator chains, 24 WMMAs per K-step of 32.
// Block: 8 waves -> 256 x 64 tile. LDA/LDB/KDIM are compile-time so subtile
// addressing folds into the 24-bit instruction offset (4 live pointers total).
// OUT: 0 = fp32 C; 1 = split bf16 (Ch,Cl) row-major; 2 = split bf16 transposed [N x M].
template<int OUT, int LDA, int LDB, int KDIM>
__global__ __launch_bounds__(256) void gemm_bf16x3(
    const __bf16* __restrict__ Ah, const __bf16* __restrict__ Al, long strideA,
    const __bf16* __restrict__ Bh, const __bf16* __restrict__ Bl, long strideB,
    float* __restrict__ Cf, __bf16* __restrict__ Ch, __bf16* __restrict__ Cl,
    long strideC, int ldc, float scale)
{
    const int lane = threadIdx.x & 31;
    const int wave = threadIdx.x >> 5;
    const int row0 = blockIdx.y * 256 + wave * 32;
    const int col0 = blockIdx.x * 64;
    const long b  = blockIdx.z;

    const int r  = lane & 15;
    const int hh = lane >> 4;
    const int n  = lane & 15;

    // Only 4 live pointers; all m/s subtile deltas are compile-time immediates.
    const __bf16* pAh = Ah + b * strideA + (long)(row0 + r) * LDA + hh * 16;
    const __bf16* pAl = Al + b * strideA + (long)(row0 + r) * LDA + hh * 16;
    const __bf16* pBh = Bh + b * strideB + (long)(col0 + n) * LDB + hh * 16;
    const __bf16* pBl = Bl + b * strideB + (long)(col0 + n) * LDB + hh * 16;

    v8f acc[2][4] = {};
    for (int k0 = 0; k0 < KDIM; k0 += 32) {
        __builtin_prefetch(pAh + 64, 0, 0);
        __builtin_prefetch(pBh + 64, 0, 0);

        v16bf ah[2], al[2];
#pragma unroll
        for (int m = 0; m < 2; ++m) {
            ah[m] = load_a_frag(pAh + m * 16 * LDA);
            al[m] = load_a_frag(pAl + m * 16 * LDA);
        }
        v16bf bh[4], bl[4];
#pragma unroll
        for (int s = 0; s < 4; ++s) {
            bh[s] = *(const v16bf*)(pBh + s * 16 * LDB);
            bl[s] = *(const v16bf*)(pBl + s * 16 * LDB);
        }

        // Product-outermost issue order: dependent WMMAs on the same accumulator
        // are 8 issues apart -> clears the 5-slot bf16 WMMA hazard window.
#pragma unroll
        for (int p = 0; p < 3; ++p) {
#pragma unroll
            for (int s = 0; s < 4; ++s) {
#pragma unroll
                for (int m = 0; m < 2; ++m) {
                    const v16bf& a = (p == 2) ? al[m] : ah[m];
                    const v16bf& bb = (p == 1) ? bl[s] : bh[s];
                    acc[m][s] = __builtin_amdgcn_wmma_f32_16x16x32_bf16(
                        false, a, false, bb, (short)0, acc[m][s], false, false);
                }
            }
        }

        pAh += 32; pAl += 32; pBh += 32; pBl += 32;
    }

    // C/D layout: lane -> N = lane&15 ; VGPR v -> M = v + 8*(lane>>4)
#pragma unroll
    for (int m = 0; m < 2; ++m) {
#pragma unroll
        for (int s = 0; s < 4; ++s) {
            const int cc = col0 + s * 16 + n;
#pragma unroll
            for (int v = 0; v < 8; ++v) {
                const int rr = row0 + m * 16 + 8 * hh + v;
                float f = acc[m][s][v] * scale;
                if (OUT == 0) {
                    Cf[b * strideC + (long)rr * ldc + cc] = f;
                } else if (OUT == 1) {
                    __bf16 h, l; split_bf16(f, h, l);
                    long idx = b * strideC + (long)rr * ldc + cc;
                    Ch[idx] = h; Cl[idx] = l;
                } else {
                    __bf16 h, l; split_bf16(f, h, l);
                    long idx = b * strideC + (long)cc * ldc + rr;  // transposed store
                    Ch[idx] = h; Cl[idx] = l;
                }
            }
        }
    }
}

__global__ __launch_bounds__(256) void split_x_kernel(
    const float* __restrict__ x, __bf16* __restrict__ xh, __bf16* __restrict__ xl, long n)
{
    long i = (long)blockIdx.x * 256 + threadIdx.x;
    if (i < n) {
        __bf16 h, l; split_bf16(x[i], h, l);
        xh[i] = h; xl[i] = l;
    }
}

// w: [4][DIM][DIM]; out wt[s][e][d] = split(w[s][d][e]) for s in 0..2
__global__ __launch_bounds__(256) void split_w_transpose_kernel(
    const float* __restrict__ w, __bf16* __restrict__ wth, __bf16* __restrict__ wtl)
{
    int e = blockIdx.x * 256 + threadIdx.x;
    int d = blockIdx.y;
    int s = blockIdx.z;
    float f = w[((long)s * DIM + d) * DIM + e];
    __bf16 h, l; split_bf16(f, h, l);
    long o = ((long)s * DIM + e) * DIM + d;
    wth[o] = h; wtl[o] = l;
}

// One 256-thread block per row of S [BATCH*SEQ rows, SEQ cols].
// In-place rewrite: row's 2048 fp32 -> 2048 bf16 (hi) followed by 2048 bf16 (lo).
__global__ __launch_bounds__(256) void softmax_split_kernel(float* __restrict__ S)
{
    const long row = blockIdx.x;
    float* rowp = S + row * (long)SEQ;
    const int t = threadIdx.x;
    __shared__ float red[256];

    float v[8];
#pragma unroll
    for (int i = 0; i < 8; ++i) v[i] = rowp[t + 256 * i];

    float m = v[0];
#pragma unroll
    for (int i = 1; i < 8; ++i) m = fmaxf(m, v[i]);
    red[t] = m;
    __syncthreads();
    for (int s = 128; s > 0; s >>= 1) {
        if (t < s) red[t] = fmaxf(red[t], red[t + s]);
        __syncthreads();
    }
    m = red[0];
    __syncthreads();

    float sum = 0.0f;
#pragma unroll
    for (int i = 0; i < 8; ++i) { v[i] = __expf(v[i] - m); sum += v[i]; }
    red[t] = sum;
    __syncthreads();
    for (int s = 128; s > 0; s >>= 1) {
        if (t < s) red[t] += red[t + s];
        __syncthreads();
    }
    const float inv = 1.0f / red[0];

    __bf16* outh = (__bf16*)rowp;   // first 4 KB of the row
    __bf16* outl = outh + SEQ;      // second 4 KB
#pragma unroll
    for (int i = 0; i < 8; ++i) {
        float p = v[i] * inv;
        __bf16 h, l; split_bf16(p, h, l);
        outh[t + 256 * i] = h;
        outl[t + 256 * i] = l;
    }
}

extern "C" void kernel_launch(void* const* d_in, const int* in_sizes, int n_in,
                              void* d_out, int out_size, void* d_ws, size_t ws_size,
                              hipStream_t stream) {
    const float* x = (const float*)d_in[0];   // [4][2048][1024]
    const float* w = (const float*)d_in[1];   // [4][1024][1024]
    float* out = (float*)d_out;               // [4][2048][1024]
    char* ws = (char*)d_ws;

    const size_t MB = 1024ull * 1024ull;
    // Region 0 (64 MB): X/W splits during projections, then reused for S (softmax in place).
    __bf16* Xh  = (__bf16*)(ws + 0 * MB);
    __bf16* Xl  = (__bf16*)(ws + 16 * MB);
    __bf16* Wth = (__bf16*)(ws + 32 * MB);
    __bf16* Wtl = (__bf16*)(ws + 38 * MB);
    float*  Smat = (float*)(ws + 0 * MB);
    // Region 1 (96 MB): split-bf16 Q, K, V^T
    __bf16* Qh  = (__bf16*)(ws + 64 * MB);
    __bf16* Ql  = (__bf16*)(ws + 80 * MB);
    __bf16* Kh  = (__bf16*)(ws + 96 * MB);
    __bf16* Kl  = (__bf16*)(ws + 112 * MB);
    __bf16* Vth = (__bf16*)(ws + 128 * MB);
    __bf16* Vtl = (__bf16*)(ws + 144 * MB);

    const long nX = (long)BATCH * SEQ * DIM;
    split_x_kernel<<<dim3((unsigned)(nX / 256)), 256, 0, stream>>>(x, Xh, Xl, nX);
    split_w_transpose_kernel<<<dim3(DIM / 256, DIM, 3), 256, 0, stream>>>(w, Wth, Wtl);

    const long sXD = (long)SEQ * DIM;   // batch stride for X/Q/K/V
    const long sSS = (long)SEQ * SEQ;   // batch stride for S (fp32 elems)
    const long wSl = (long)DIM * DIM;   // weight slice stride

    // --- Projections: [SEQ x DIM] = X @ W, per batch ---
    dim3 gProj(DIM / 64, SEQ / 256, BATCH);
    // Q (scale 1/sqrt(D_OUT) = 1/32 folded in)
    gemm_bf16x3<1, DIM, DIM, DIM><<<gProj, 256, 0, stream>>>(
        Xh, Xl, sXD, Wth + 0 * wSl, Wtl + 0 * wSl, 0L,
        nullptr, Qh, Ql, sXD, DIM, 0.03125f);
    // K
    gemm_bf16x3<1, DIM, DIM, DIM><<<gProj, 256, 0, stream>>>(
        Xh, Xl, sXD, Wth + 1 * wSl, Wtl + 1 * wSl, 0L,
        nullptr, Kh, Kl, sXD, DIM, 1.0f);
    // V, stored transposed as Vt[e][s] so AV GEMM's B-operand loads are contiguous
    gemm_bf16x3<2, DIM, DIM, DIM><<<gProj, 256, 0, stream>>>(
        Xh, Xl, sXD, Wth + 2 * wSl, Wtl + 2 * wSl, 0L,
        nullptr, Vth, Vtl, sXD, SEQ, 1.0f);

    // --- S = Q @ K^T : B^T == K row-major, so pass K directly ---
    dim3 gS(SEQ / 64, SEQ / 256, BATCH);
    gemm_bf16x3<0, DIM, DIM, DIM><<<gS, 256, 0, stream>>>(
        Qh, Ql, sXD, Kh, Kl, sXD,
        Smat, nullptr, nullptr, sSS, SEQ, 1.0f);

    // --- softmax rows, rewrite in place as (hi,lo) bf16 planes per row ---
    softmax_split_kernel<<<dim3(BATCH * SEQ), 256, 0, stream>>>(Smat);

    // --- O = A @ V : A rows are bf16 hi@+0 / lo@+SEQ, row stride 2*SEQ ---
    dim3 gO(DIM / 64, SEQ / 256, BATCH);
    gemm_bf16x3<0, 2 * SEQ, SEQ, SEQ><<<gO, 256, 0, stream>>>(
        (const __bf16*)Smat, (const __bf16*)Smat + SEQ, sSS * 2,
        Vth, Vtl, sXD,
        out, nullptr, nullptr, sXD, DIM, 1.0f);
}
